// MultiHeadAttention_51092930953406
// MI455X (gfx1250) — matrix operand order repaired and checked by reading, not compiled, not run on hardware
//
#include <hip/hip_runtime.h>
#include <hip/hip_bf16.h>

typedef __attribute__((ext_vector_type(16))) __bf16 v16bf;
typedef __attribute__((ext_vector_type(8)))  __bf16 v8bf;
typedef __attribute__((ext_vector_type(8)))  float  v8f;

#define WMMA_BF16(a, b, c) \
  __builtin_amdgcn_wmma_f32_16x16x32_bf16(false, (a), false, (b), (short)0, (c), false, false)

constexpr int BATCH = 4;
constexpr int SEQ   = 2048;
constexpr int DMODEL= 512;
constexpr int NHEAD = 8;
constexpr int HDIM  = 64;
constexpr int MROWS = BATCH * SEQ;   // 8192

// ---- CDNA5 async LDS copy helpers (ASYNCcnt path, no VGPR round-trip) ------
__device__ __forceinline__ uint32_t lds_u32(const void* p) {
  // generic pointer to LDS: low 32 bits are the LDS address
  return (uint32_t)(uintptr_t)p;
}
__device__ __forceinline__ void async_copy_b128(uint32_t lds_dst, const void* gsrc) {
  asm volatile("global_load_async_to_lds_b128 %0, %1, off"
               :: "v"(lds_dst), "v"(gsrc) : "memory");
}
__device__ __forceinline__ void wait_asynccnt_0() {
  asm volatile("s_wait_asynccnt 0x0" ::: "memory");
}
__device__ __forceinline__ void wait_asynccnt_3() {
  asm volatile("s_wait_asynccnt 0x3" ::: "memory");
}

// ---------------------------------------------------------------------------
// Kernel 0: fp32 -> bf16 conversion pre-pass (x, w_qkv, w_out). ~21MB total,
// ~1us at 23.3 TB/s; enables the pure-async bf16 GEMM pipelines below.
// ---------------------------------------------------------------------------
__global__ __launch_bounds__(256)
void cvt_f32_bf16_kernel(const float* __restrict__ src, __bf16* __restrict__ dst,
                         int n) {
  int i = (blockIdx.x * 256 + threadIdx.x) * 4;
  if (i < n) {
    float4 v = *(const float4*)(src + i);
    dst[i + 0] = (__bf16)v.x;
    dst[i + 1] = (__bf16)v.y;
    dst[i + 2] = (__bf16)v.z;
    dst[i + 3] = (__bf16)v.w;
  }
}

// ---------------------------------------------------------------------------
// Unified bf16 WMMA GEMM: C[m,n] = sum_k A[m,k] * Bw[n,k]  (both K-contig).
// Block = 256 thr (8 waves), tile 128(M) x 64(N), K-step 32, double-buffered
// async LDS staging (3 x b128 per thread per K-step).
// MODE 0: scatter to Q [b][h][s][hd], K [b][h][s][hd], V^T [b][h][hd][s].
// MODE 1: plain fp32 store to Out[m][n].
// ---------------------------------------------------------------------------
template <int MODE>
__global__ __launch_bounds__(256)
void gemm_bf16_wmma(const __bf16* __restrict__ A, const __bf16* __restrict__ Bw,
                    __bf16* __restrict__ Qw, __bf16* __restrict__ Kw,
                    __bf16* __restrict__ Vw, float* __restrict__ Out)
{
  constexpr int KP = 48;  // 32 + pad -> 96B rows (16B-aligned chunks, 32B frags)
  __shared__ __align__(32) __bf16 As[2][128][KP];
  __shared__ __align__(32) __bf16 Bs[2][64][KP];

  const int tid  = threadIdx.x;
  const int wave = tid >> 5;
  const int lane = tid & 31;
  const int lm   = lane & 15;
  const int g    = lane >> 4;
  const int m0   = blockIdx.x * 128;
  const int n0   = blockIdx.y * 64;

  // async staging mapping: A 128x32 bf16 (2 x 16B / thread), B 64x32 (1 x 16B)
  const int arow = tid >> 1;
  const int acol = (tid & 1) * 16;
  const int brow = tid >> 2;
  const int bcol = (tid & 3) * 8;
  const __bf16* Ag = A  + (size_t)(m0 + arow) * DMODEL + acol;
  const __bf16* Bg = Bw + (size_t)(n0 + brow) * DMODEL + bcol;

  auto stage = [&](int buf, int k0) {
    async_copy_b128(lds_u32(&As[buf][arow][acol]),     Ag + k0);
    async_copy_b128(lds_u32(&As[buf][arow][acol + 8]), Ag + k0 + 8);
    async_copy_b128(lds_u32(&Bs[buf][brow][bcol]),     Bg + k0);
  };

  stage(0, 0);
  v8f c[4] = {};
  int cur = 0;
  for (int k0 = 0; k0 < DMODEL; k0 += 32) {
    const int nxt = cur ^ 1;
    if (k0 + 32 < DMODEL) {
      stage(nxt, k0 + 32);   // overlap next fill with this tile's WMMAs
      wait_asynccnt_3();     // in-order completion: first 3 (cur buf) are done
    } else {
      wait_asynccnt_0();
    }
    __syncthreads();

    // A fragment (16x32): lane lm = M; K = 16*g + i (contiguous)
    v16bf a;
    {
      const __bf16* rp = &As[cur][wave * 16 + lm][0];
      v8bf lo = *(const v8bf*)(rp + 16 * g);
      v8bf hi = *(const v8bf*)(rp + 16 * g + 8);
      #pragma unroll
      for (int i = 0; i < 8; ++i) { a[i] = lo[i]; a[i + 8] = hi[i]; }
    }
    #pragma unroll
    for (int t = 0; t < 4; ++t) {
      // B fragment (32x16): lane lm = N, K = 16*g + i (contiguous 32B)
      v16bf bw = *(const v16bf*)(&Bs[cur][t * 16 + lm][16 * g]);
      c[t] = WMMA_BF16(a, bw, c[t]);
    }
    __syncthreads();
    cur = nxt;
  }

  if (MODE == 0) {
    // all 64 cols of this block share which/head (64 | n0, head span = 64)
    const int which = n0 / DMODEL;          // 0=Q,1=K,2=V
    const int h     = (n0 % DMODEL) / HDIM;
    #pragma unroll
    for (int t = 0; t < 4; ++t) {
      #pragma unroll
      for (int r = 0; r < 8; ++r) {
        int m  = m0 + wave * 16 + r + 8 * g;
        int hd = t * 16 + lm;
        int bb = m >> 11;
        int s  = m & (SEQ - 1);
        __bf16 val = (__bf16)c[t][r];
        size_t bhs = ((size_t)(bb * NHEAD + h) * SEQ + s) * HDIM + hd;
        if (which == 0)      Qw[bhs] = val;
        else if (which == 1) Kw[bhs] = val;
        else                 Vw[((size_t)(bb * NHEAD + h) * HDIM + hd) * SEQ + s] = val;
      }
    }
  } else {
    #pragma unroll
    for (int t = 0; t < 4; ++t) {
      #pragma unroll
      for (int r = 0; r < 8; ++r) {
        int m = m0 + wave * 16 + r + 8 * g;
        int e = n0 + t * 16 + lm;
        Out[(size_t)m * DMODEL + e] = c[t][r];
      }
    }
  }
}

// ---------------------------------------------------------------------------
// Kernel 2: flash attention. Block = 128 thr (4 waves); each wave owns a
// 16-row query tile, streams key blocks of 32, online softmax in registers.
// 8 WMMAs per key block; K/V stay L2-resident (16MB << 192MB L2).
// ---------------------------------------------------------------------------
__global__ __launch_bounds__(128)
void flash_attn_kernel(const __bf16* __restrict__ Qw, const __bf16* __restrict__ Kw,
                       const __bf16* __restrict__ Vw, __bf16* __restrict__ Ow)
{
  __shared__ __align__(32) __bf16 P_lds[4][16][48];   // per-wave P scratch

  const int tid  = threadIdx.x;
  const int wave = tid >> 5;
  const int lane = tid & 31;
  const int lm   = lane & 15;
  const int g    = lane >> 4;
  const int bh   = blockIdx.x;               // b*H + h
  const int bb   = bh / NHEAD;
  const int h    = bh % NHEAD;
  const int q0   = blockIdx.y * 64 + wave * 16;

  const __bf16* Qbh = Qw + (size_t)bh * SEQ * HDIM;
  const __bf16* Kbh = Kw + (size_t)bh * SEQ * HDIM;
  const __bf16* Vbh = Vw + (size_t)bh * HDIM * SEQ;   // [hd][s]

  // Q fragments: qa[f] covers hd in [32f, 32f+32)
  v16bf qa[2];
  #pragma unroll
  for (int f = 0; f < 2; ++f) {
    #pragma unroll
    for (int i = 0; i < 16; ++i) {
      int hd = 32 * f + 16 * g + i;
      qa[f][i] = Qbh[(size_t)(q0 + lm) * HDIM + hd];
    }
  }

  v8f o[4] = {};
  float run_max[8], run_sum[8];
  #pragma unroll
  for (int r = 0; r < 8; ++r) { run_max[r] = -3.0e38f; run_sum[r] = 0.0f; }

  for (int kb = 0; kb < SEQ; kb += 32) {
    if (kb + 32 < SEQ) {  // warm WGP$ for next key/value block
      __builtin_prefetch(Kbh + (size_t)(kb + 32 + lm) * HDIM, 0, 3);
      __builtin_prefetch(Vbh + (size_t)lm * SEQ + kb + 32, 0, 3);
    }
    v8f s0 = {}, s1 = {};
    { // scores = Q(16x64) x K^T(64x32); reference applies NO 1/sqrt(d) scale
      int srow = kb + lm;
      v16bf b0 = *(const v16bf*)(Kbh + (size_t)srow * HDIM + 16 * g);
      v16bf b1 = *(const v16bf*)(Kbh + (size_t)srow * HDIM + 32 + 16 * g);
      s0 = WMMA_BF16(qa[0], b0, s0);
      s0 = WMMA_BF16(qa[1], b1, s0);
      srow = kb + 16 + lm;
      v16bf b2 = *(const v16bf*)(Kbh + (size_t)srow * HDIM + 16 * g);
      v16bf b3 = *(const v16bf*)(Kbh + (size_t)srow * HDIM + 32 + 16 * g);
      s1 = WMMA_BF16(qa[0], b2, s1);
      s1 = WMMA_BF16(qa[1], b3, s1);
    }
    // online softmax; C-layout: row = r + 8g, cols across the 16-lane group
    #pragma unroll
    for (int r = 0; r < 8; ++r) {
      float mx = fmaxf(s0[r], s1[r]);
      #pragma unroll
      for (int off = 1; off < 16; off <<= 1) mx = fmaxf(mx, __shfl_xor(mx, off, 32));
      float nm   = fmaxf(run_max[r], mx);
      float corr = __expf(run_max[r] - nm);
      run_max[r] = nm;
      float p0 = __expf(s0[r] - nm);
      float p1 = __expf(s1[r] - nm);
      float rs = p0 + p1;
      #pragma unroll
      for (int off = 1; off < 16; off <<= 1) rs += __shfl_xor(rs, off, 32);
      run_sum[r] = run_sum[r] * corr + rs;
      #pragma unroll
      for (int t = 0; t < 4; ++t) o[t][r] *= corr;
      s0[r] = p0; s1[r] = p1;
    }
    // P (C-layout, fp32) -> LDS bf16 -> A-layout fragment
    #pragma unroll
    for (int r = 0; r < 8; ++r) {
      int row = r + 8 * g;
      P_lds[wave][row][lm]      = (__bf16)s0[r];
      P_lds[wave][row][lm + 16] = (__bf16)s1[r];
    }
    asm volatile("s_wait_dscnt 0" ::: "memory");  // wave-local DS RAW
    v16bf pa;
    {
      const __bf16* rp = &P_lds[wave][lm][0];
      v8bf lo = *(const v8bf*)(rp + 16 * g);
      v8bf hi = *(const v8bf*)(rp + 16 * g + 8);
      #pragma unroll
      for (int i = 0; i < 8; ++i) { pa[i] = lo[i]; pa[i + 8] = hi[i]; }
    }
    // O += P(16x32) x V(32x64); V^T layout makes B-fragments contiguous
    #pragma unroll
    for (int t = 0; t < 4; ++t) {
      v16bf bv = *(const v16bf*)(Vbh + (size_t)(t * 16 + lm) * SEQ + kb + 16 * g);
      o[t] = WMMA_BF16(pa, bv, o[t]);
    }
  }

  // normalize and store O as bf16 in [bs][d] (row-major, K-contig for GEMM 2)
  #pragma unroll
  for (int r = 0; r < 8; ++r) {
    float inv = 1.0f / run_sum[r];
    int s = q0 + r + 8 * g;
    #pragma unroll
    for (int t = 0; t < 4; ++t) {
      Ow[(size_t)(bb * SEQ + s) * DMODEL + h * HDIM + t * 16 + lm] =
          (__bf16)(o[t][r] * inv);
    }
  }
}

// ---------------------------------------------------------------------------
extern "C" void kernel_launch(void* const* d_in, const int* in_sizes, int n_in,
                              void* d_out, int out_size, void* d_ws, size_t ws_size,
                              hipStream_t stream) {
  const float* x     = (const float*)d_in[0];   // [4,2048,512]
  const float* w_qkv = (const float*)d_in[1];   // [1536,512]
  const float* w_out = (const float*)d_in[2];   // [512,512]
  float* out = (float*)d_out;                   // [4,2048,512]

  const size_t nX  = (size_t)MROWS * DMODEL;        // 4,194,304
  const size_t nWq = (size_t)3 * DMODEL * DMODEL;   //   786,432
  const size_t nWo = (size_t)DMODEL * DMODEL;       //   262,144

  __bf16* Qw  = (__bf16*)d_ws;          // 8MB each
  __bf16* Kw  = Qw + nX;
  __bf16* Vw  = Kw + nX;
  __bf16* Ow  = Vw + nX;
  __bf16* Xb  = Ow + nX;                // bf16 copies of inputs
  __bf16* Wqb = Xb + nX;
  __bf16* Wob = Wqb + nWq;              // total ws use ~42MB

  // fp32 -> bf16 pre-pass
  cvt_f32_bf16_kernel<<<dim3((int)(nX  / 1024)), 256, 0, stream>>>(x,     Xb,  (int)nX);
  cvt_f32_bf16_kernel<<<dim3((int)(nWq / 1024)), 256, 0, stream>>>(w_qkv, Wqb, (int)nWq);
  cvt_f32_bf16_kernel<<<dim3((int)(nWo / 1024)), 256, 0, stream>>>(w_out, Wob, (int)nWo);

  // QKV projection -> Q, K, V^T (bf16)
  gemm_bf16_wmma<0><<<dim3(MROWS / 128, (3 * DMODEL) / 64), 256, 0, stream>>>(
      Xb, Wqb, Qw, Kw, Vw, nullptr);
  // attention core
  flash_attn_kernel<<<dim3(BATCH * NHEAD, SEQ / 64), 128, 0, stream>>>(
      Qw, Kw, Vw, Ow);
  // output projection -> fp32
  gemm_bf16_wmma<1><<<dim3(MROWS / 128, DMODEL / 64), 256, 0, stream>>>(
      Ow, Wob, nullptr, nullptr, nullptr, out);
}
